// BahdanauAttention_39015482916865
// MI455X (gfx1250) — matrix operand order repaired and checked by reading, not compiled, not run on hardware
//
#include <hip/hip_runtime.h>
#include <hip/hip_bf16.h>
#include <math.h>

// ---- problem constants (from reference) ----
#define BATCH 64
#define SEQ   2048
#define HID   1024   // H (= K of big GEMM)
#define UNITS 1024   // U (= N of big GEMM)

typedef __attribute__((ext_vector_type(16))) __bf16 v16bf;
typedef __attribute__((ext_vector_type(8)))  __bf16 v8bf;
typedef __attribute__((ext_vector_type(8)))  float  v8f;
typedef __attribute__((ext_vector_type(4)))  float  v4f;

// Hardware tanh (CDNA5 V_TANH_F32 trans op) — ocml tanhf is ~20 VALU ops and
// we evaluate it 134M times; the trans unit co-executes with WMMA.
static __device__ __forceinline__ float fast_tanh(float x) {
#if __has_builtin(__builtin_amdgcn_tanhf)
  return __builtin_amdgcn_tanhf(x);
#else
  float y;
  asm volatile("v_tanh_f32 %0, %1" : "=v"(y) : "v"(x));
  return y;
#endif
}

// ---------------------------------------------------------------------------
// Kernel 0: W2 [H][U] fp32  ->  W2T [U][H] bf16   (tiled transpose via LDS)
// ---------------------------------------------------------------------------
__global__ __launch_bounds__(256) void k_w2_transpose(
    const float* __restrict__ W2, __bf16* __restrict__ W2T) {
  __shared__ float tile[64][65];
  const int h0 = blockIdx.x * 64;
  const int u0 = blockIdx.y * 64;
  const int tx = threadIdx.x & 63;
  const int ty = threadIdx.x >> 6;            // 0..3
#pragma unroll
  for (int j = 0; j < 16; ++j) {
    const int r = ty + j * 4;                 // 0..63
    tile[r][tx] = W2[(size_t)(h0 + r) * UNITS + (u0 + tx)];
  }
  __syncthreads();
#pragma unroll
  for (int j = 0; j < 16; ++j) {
    const int r = ty + j * 4;
    W2T[(size_t)(u0 + r) * HID + (h0 + tx)] = (__bf16)tile[tx][r];
  }
}

// ---------------------------------------------------------------------------
// Kernel 1: qp[b][u] = query[b]@W1[:,u] + b1[u] + b2[u]   (tiny: 0.13 GFLOP)
// ---------------------------------------------------------------------------
__global__ __launch_bounds__(256) void k_qproj(
    const float* __restrict__ query, const float* __restrict__ W1,
    const float* __restrict__ b1, const float* __restrict__ b2,
    float* __restrict__ qp) {
  __shared__ float qrow[HID];
  const int b = blockIdx.x;
  const int u = blockIdx.y * 256 + threadIdx.x;
  for (int i = threadIdx.x; i < HID; i += 256) qrow[i] = query[b * HID + i];
  __syncthreads();
  float acc = b1[u] + b2[u];
#pragma unroll 4
  for (int h = 0; h < HID; ++h) acc = fmaf(qrow[h], W1[(size_t)h * UNITS + u], acc);
  qp[b * UNITS + u] = acc;
}

// ---------------------------------------------------------------------------
// Kernel 2: fused score GEMM (the 275 GFLOP hot spot).
//   score[row] = sum_u tanh( values[row]@W2[:,u] + qp[b][u] ) * V[u] + bv
// A tile (64 x 1024 bf16) staged once in LDS (128 KB of 320 KB/WGP);
// B fragments streamed from W2T (L2-resident, read once per workgroup).
// Wave task = one 32-col strip x 4 row tiles (rt=4, ct=2):
//   per k-step 4 A frags + 2 B frags feed 8 WMMAs -> 0.5 KB LDS/WMMA,
//   within the LDS port budget. unroll(1) caps live fragments (no spills).
// ---------------------------------------------------------------------------
__global__ __launch_bounds__(256) void k_score(
    const float* __restrict__ values, const __bf16* __restrict__ W2T,
    const float* __restrict__ qp, const float* __restrict__ V,
    const float* __restrict__ bv, float* __restrict__ scores) {
  __shared__ __bf16 As[64 * HID];     // 128 KB
  __shared__ float  s_score[64];

  const int tid = threadIdx.x;
  const int b   = blockIdx.x >> 5;          // 32 row-chunks per batch
  const int s0  = (blockIdx.x & 31) * 64;

  if (tid < 64) s_score[tid] = 0.0f;

  // ---- stage A: values[b, s0..s0+63, :] fp32 -> bf16 LDS (coalesced f32x4)
  const float* src = values + (size_t)(b * SEQ + s0) * HID;
  for (int i = tid; i < 64 * (HID / 4); i += 256) {
    v4f x = *(const v4f*)(src + (size_t)i * 4);
    __bf16* d = &As[i * 4];
    d[0] = (__bf16)x[0]; d[1] = (__bf16)x[1];
    d[2] = (__bf16)x[2]; d[3] = (__bf16)x[3];
  }
  __syncthreads();

  const int lane   = tid & 31;
  const int w      = tid >> 5;        // wave id 0..7
  const int nl     = lane & 15;       // column within 16x16 tile (C/B layout)
  const int half   = lane >> 4;       // lane-half selects K sub-ranges
  const int kbaseA = half * 16;       // A frag: contiguous 16-elem K run
  const int koffB  = half * 16;       // B frag: contiguous 16-elem K run

  float part[4][8];                   // per-row tanh-dot partials (reg accum)
#pragma unroll
  for (int rt = 0; rt < 4; ++rt)
#pragma unroll
    for (int v = 0; v < 8; ++v) part[rt][v] = 0.0f;

  // 32 strips of 32 columns; wave w handles strips w, w+8, w+16, w+24
  for (int p = 0; p < 4; ++p) {
    const int n0 = (w + 8 * p) * 32;        // strip base column (wave-uniform)
    const __bf16* bptr0 = W2T + (size_t)(n0 + nl) * HID + koffB;
    const __bf16* bptr1 = W2T + (size_t)(n0 + 16 + nl) * HID + koffB;

    v8f acc[4][2];
#pragma unroll
    for (int rt = 0; rt < 4; ++rt) {
      acc[rt][0] = (v8f)(0.0f);
      acc[rt][1] = (v8f)(0.0f);
    }

#pragma unroll 1
    for (int k0 = 0; k0 < HID; k0 += 32) {
      v16bf bfrag0 = *(const v16bf*)(bptr0 + k0);   // each reused by 4 WMMAs
      v16bf bfrag1 = *(const v16bf*)(bptr1 + k0);
#pragma unroll
      for (int rt = 0; rt < 4; ++rt) {
        const __bf16* ap = &As[(rt * 16 + nl) * HID + k0 + kbaseA];
        v8bf alo = *(const v8bf*)ap;
        v8bf ahi = *(const v8bf*)(ap + 8);
        v16bf afrag = __builtin_shufflevector(
            alo, ahi, 0, 1, 2, 3, 4, 5, 6, 7, 8, 9, 10, 11, 12, 13, 14, 15);
        acc[rt][0] = __builtin_amdgcn_wmma_f32_16x16x32_bf16(
            false, afrag, false, bfrag0, (short)0, acc[rt][0], false, false);
        acc[rt][1] = __builtin_amdgcn_wmma_f32_16x16x32_bf16(
            false, afrag, false, bfrag1, (short)0, acc[rt][1], false, false);
      }
    }

    // epilogue: one column per lane per ct -> tanh(acc + qp[b][n]) * V[n]
#pragma unroll
    for (int ct = 0; ct < 2; ++ct) {
      const int n    = n0 + ct * 16 + nl;
      const float q  = qp[b * UNITS + n];
      const float vn = V[n];
#pragma unroll
      for (int rt = 0; rt < 4; ++rt)
#pragma unroll
        for (int v = 0; v < 8; ++v)
          part[rt][v] = fmaf(fast_tanh(acc[rt][ct][v] + q), vn, part[rt][v]);
    }
  }

  // cross-lane / cross-wave row reduction (ds_add_f32)
#pragma unroll
  for (int rt = 0; rt < 4; ++rt)
#pragma unroll
    for (int v = 0; v < 8; ++v)
      atomicAdd(&s_score[rt * 16 + half * 8 + v], part[rt][v]);
  __syncthreads();

  if (tid < 64) scores[b * SEQ + s0 + tid] = s_score[tid] + bv[0];
}

// ---------------------------------------------------------------------------
// Kernel 3: softmax over S per batch -> attention weights (output region 2)
// ---------------------------------------------------------------------------
__global__ __launch_bounds__(256) void k_softmax(
    const float* __restrict__ scores, float* __restrict__ wts) {
  __shared__ float red[256];
  const int b = blockIdx.x;
  const int tid = threadIdx.x;
  const float* srow = scores + b * SEQ;
  float x[8];
  float lmax = -INFINITY;
#pragma unroll
  for (int i = 0; i < 8; ++i) {
    x[i] = srow[tid + i * 256];
    lmax = fmaxf(lmax, x[i]);
  }
  red[tid] = lmax;
  __syncthreads();
  for (int off = 128; off > 0; off >>= 1) {
    if (tid < off) red[tid] = fmaxf(red[tid], red[tid + off]);
    __syncthreads();
  }
  const float m = red[0];
  __syncthreads();
  float lsum = 0.0f;
#pragma unroll
  for (int i = 0; i < 8; ++i) {
    x[i] = expf(x[i] - m);
    lsum += x[i];
  }
  red[tid] = lsum;
  __syncthreads();
  for (int off = 128; off > 0; off >>= 1) {
    if (tid < off) red[tid] += red[tid + off];
    __syncthreads();
  }
  const float inv = 1.0f / red[0];
#pragma unroll
  for (int i = 0; i < 8; ++i) wts[b * SEQ + tid + i * 256] = x[i] * inv;
}

// ---------------------------------------------------------------------------
// Kernel 4: context[b][h] = sum_s w[b][s] * values[b][s][h]   (memory-bound,
// float4 per lane: each block streams full 4 KB rows, coalesced)
// ---------------------------------------------------------------------------
__global__ __launch_bounds__(256) void k_context(
    const float* __restrict__ values, const float* __restrict__ wts,
    float* __restrict__ ctx) {
  __shared__ float wrow[SEQ];         // 8 KB
  const int b = blockIdx.x;
  for (int i = threadIdx.x; i < SEQ; i += 256) wrow[i] = wts[b * SEQ + i];
  __syncthreads();
  const float* vb = values + (size_t)b * SEQ * HID + threadIdx.x * 4;
  v4f acc = (v4f)(0.0f);
#pragma unroll 4
  for (int s = 0; s < SEQ; ++s) {
    const v4f x = *(const v4f*)(vb + (size_t)s * HID);
    acc += x * wrow[s];
  }
  *(v4f*)(ctx + b * HID + threadIdx.x * 4) = acc;
}

// ---------------------------------------------------------------------------
// launch
// ---------------------------------------------------------------------------
extern "C" void kernel_launch(void* const* d_in, const int* in_sizes, int n_in,
                              void* d_out, int out_size, void* d_ws, size_t ws_size,
                              hipStream_t stream) {
  const float* query  = (const float*)d_in[0];
  const float* values = (const float*)d_in[1];
  const float* W1     = (const float*)d_in[2];
  const float* b1     = (const float*)d_in[3];
  const float* W2     = (const float*)d_in[4];
  const float* b2     = (const float*)d_in[5];
  const float* V      = (const float*)d_in[6];
  const float* bv     = (const float*)d_in[7];

  // workspace layout (bytes): [0,2MB) W2T bf16; [2MB,2.25MB) qp; [2.25MB,..) scores
  char* ws = (char*)d_ws;
  __bf16* W2T    = (__bf16*)(ws);
  float*  qp     = (float*)(ws + (size_t)2 * 1024 * 1024);
  float*  scores = (float*)(ws + (size_t)2 * 1024 * 1024 + 256 * 1024);

  float* ctx = (float*)d_out;                 // [B,HID]  first output
  float* wts = (float*)d_out + BATCH * HID;   // [B,SEQ,1] second output

  k_w2_transpose<<<dim3(HID / 64, UNITS / 64), 256, 0, stream>>>(W2, W2T);
  k_qproj<<<dim3(BATCH, UNITS / 256), 256, 0, stream>>>(query, W1, b1, b2, qp);
  k_score<<<dim3(BATCH * (SEQ / 64)), 256, 0, stream>>>(values, W2T, qp, V, bv, scores);
  k_softmax<<<dim3(BATCH), 256, 0, stream>>>(scores, wts);
  k_context<<<dim3(BATCH), 256, 0, stream>>>(values, wts, ctx);
}
